// KPNLPnetwork_3487513445091
// MI455X (gfx1250) — matrix operand order repaired and checked by reading, not compiled, not run on hardware
//
#include <hip/hip_runtime.h>

typedef _Float16 half_t;
typedef __attribute__((ext_vector_type(16))) _Float16 v16h;
typedef __attribute__((ext_vector_type(8)))  float    v8f;

#ifndef __has_builtin
#define __has_builtin(x) 0
#endif

#if __has_builtin(__builtin_amdgcn_global_load_async_to_lds_b128)
#define HAVE_ASYNC_LDS 1
// Toolchain declares the async-to-LDS builtins with generic `v4i*` params.
typedef int avi4 __attribute__((vector_size(4 * sizeof(int))));
#else
#define HAVE_ASYNC_LDS 0
#endif

// ---------------------------------------------------------------------------
// Pack OIHW fp32 weights into WMMA B-fragment layout (fp16), plus padded bias.
// Fragment index: (((tap*KS + ks)*NT + nt)*32 + lane)*16 + h
// B k order chosen here (contiguous, matches A gather): lane<16 -> N=lane,   K=h     (0..15)
//                            lane>=16 -> N=lane-16, K=16+h (16..31)
// ---------------------------------------------------------------------------
__global__ void pack_weights_k(const float* __restrict__ w, const float* __restrict__ b,
                               half_t* __restrict__ frag, float* __restrict__ bpad,
                               int cin, int cout, int KS, int NT, int total) {
  int idx = blockIdx.x * blockDim.x + threadIdx.x;
  if (idx < 64) bpad[idx] = (idx < cout) ? b[idx] : 0.f;
  if (idx >= total) return;
  int h    = idx & 15;
  int lane = (idx >> 4) & 31;
  int rest = idx >> 9;
  int nt  = rest % NT; rest /= NT;
  int ks  = rest % KS;
  int tap = rest / KS;
  int n = nt * 16 + (lane & 15);
  int K = ks * 32 + ((lane >> 4) * 16 + h);
  int dy = tap / 3, dx = tap % 3;
  float v = 0.f;
  if (K < cin && n < cout)
    v = w[(((size_t)n * cin + K) * 3 + dy) * 3 + dx];
  frag[idx] = (half_t)v;
}

// ---------------------------------------------------------------------------
// 3x3 conv (pad 1) as implicit GEMM via 9 shifted WMMAs.
// Activations fp16 NHWC [B][H][W][CPAD_IN]; output fp16 NHWC [B][H][W][CPAD_OUT].
// Block = 256 threads = 8 wave32; tile = 16x16 pixels x (NT_OUT*16) cout.
// Each wave: 2 pixel rows (m-tiles) x NT_OUT n-tiles.
// Staging: async direct-to-LDS when available.  Epilogue: LDS transpose then
// fully coalesced b128 stores.
// ---------------------------------------------------------------------------
template <int CPAD_IN, int NT_OUT, bool LEAKY, int CPAD_OUT>
__global__ __launch_bounds__(256) void conv3x3_wmma_k(
    const half_t* __restrict__ in, const half_t* __restrict__ wfrag,
    const float* __restrict__ bias, half_t* __restrict__ out, int H, int W) {
  constexpr int KS = CPAD_IN / 32;
  constexpr int IN_ELEMS  = 18 * 18 * CPAD_IN;
  constexpr int OUT_ELEMS = 16 * 16 * CPAD_OUT;
  constexpr int SMEM_ELEMS = IN_ELEMS > OUT_ELEMS ? IN_ELEMS : OUT_ELEMS;
  __shared__ half_t smem[SMEM_ELEMS];

  const int tid = threadIdx.x;
  const int x0 = blockIdx.x * 16, y0 = blockIdx.y * 16, b = blockIdx.z;

  // ---- stage 18x18 halo tile (zero-padded borders) into LDS ----
  for (int s = tid; s < 18 * 18; s += 256) {
    int sy = s / 18, sx = s % 18;
    int gy = y0 + sy - 1, gx = x0 + sx - 1;
    half_t* dst = &smem[s * CPAD_IN];
    if (gy >= 0 && gy < H && gx >= 0 && gx < W) {
      const half_t* src = &in[((size_t)(b * H + gy) * W + gx) * CPAD_IN];
#if HAVE_ASYNC_LDS
#pragma unroll
      for (int c = 0; c < CPAD_IN / 8; ++c)
        __builtin_amdgcn_global_load_async_to_lds_b128(
            (avi4*)(src + c * 8), (avi4*)(dst + c * 8), 0, 0);
#else
      uint4* d4 = (uint4*)dst;
      const uint4* s4 = (const uint4*)src;
#pragma unroll
      for (int c = 0; c < CPAD_IN / 8; ++c) d4[c] = s4[c];
#endif
    } else {
      uint4 z = {0u, 0u, 0u, 0u};
      uint4* d4 = (uint4*)dst;
#pragma unroll
      for (int c = 0; c < CPAD_IN / 8; ++c) d4[c] = z;
    }
  }
#if HAVE_ASYNC_LDS
#if __has_builtin(__builtin_amdgcn_s_wait_asynccnt)
  __builtin_amdgcn_s_wait_asynccnt(0);
#else
  asm volatile("s_wait_asynccnt 0x0" ::: "memory");
#endif
#endif
  __syncthreads();

  const int wave = tid >> 5, lane = tid & 31;
  const int px = lane & 15, ksel = lane >> 4;
  const int py0 = wave * 2, py1 = wave * 2 + 1;

  v8f acc0[NT_OUT], acc1[NT_OUT];
#pragma unroll
  for (int nt = 0; nt < NT_OUT; ++nt) {
    v8f z = {0.f, 0.f, 0.f, 0.f, 0.f, 0.f, 0.f, 0.f};
    acc0[nt] = z; acc1[nt] = z;
  }

  __builtin_prefetch(wfrag, 0, 1);  // global_prefetch_b8 on the weight fragments

  union AB { v16h v; uint4 q[2]; };

  for (int tap = 0; tap < 9; ++tap) {
    const int dy = tap / 3, dx = tap % 3;
#pragma unroll
    for (int ks = 0; ks < KS; ++ks) {
      // A fragment gather (contiguous k order, same as B packing):
      // lane<16: halves = K{0..15}; lane>=16: halves = K{16..31}
      AB a0, a1;
      const half_t* s0 = &smem[((py0 + dy) * 18 + (px + dx)) * CPAD_IN + ks * 32 + ksel * 16];
      const half_t* s1 = &smem[((py1 + dy) * 18 + (px + dx)) * CPAD_IN + ks * 32 + ksel * 16];
      a0.q[0] = *(const uint4*)(s0);      a0.q[1] = *(const uint4*)(s0 + 8);
      a1.q[0] = *(const uint4*)(s1);      a1.q[1] = *(const uint4*)(s1 + 8);
#pragma unroll
      for (int nt = 0; nt < NT_OUT; ++nt) {
        AB bw;
        const half_t* wp = wfrag + ((((size_t)tap * KS + ks) * NT_OUT + nt) * 32 + lane) * 16;
        bw.q[0] = *(const uint4*)(wp);
        bw.q[1] = *(const uint4*)(wp + 8);
        acc0[nt] = __builtin_amdgcn_wmma_f32_16x16x32_f16(
            false, a0.v, false, bw.v, (short)0, acc0[nt], false, false);
        acc1[nt] = __builtin_amdgcn_wmma_f32_16x16x32_f16(
            false, a1.v, false, bw.v, (short)0, acc1[nt], false, false);
      }
    }
  }

  // ---- epilogue: bias + optional leaky, transpose through LDS, coalesced store
  // D layout: VGPR r holds element (M = r + 8*(lane>=16), N = lane&15).
  __syncthreads();  // smem tile fully consumed; safe to reuse as output tile
#pragma unroll
  for (int nt = 0; nt < NT_OUT; ++nt) {
    const int ch = nt * 16 + (lane & 15);
    const float bv = bias[ch];
#pragma unroll
    for (int r = 0; r < 8; ++r) {
      const int sx = r + ksel * 8;
      float v0 = acc0[nt][r] + bv;
      float v1 = acc1[nt][r] + bv;
      if (LEAKY) {
        v0 = v0 >= 0.f ? v0 : 0.01f * v0;
        v1 = v1 >= 0.f ? v1 : 0.01f * v1;
      }
      smem[(py0 * 16 + sx) * CPAD_OUT + ch] = (half_t)v0;
      smem[(py1 * 16 + sx) * CPAD_OUT + ch] = (half_t)v1;
    }
  }
  __syncthreads();

  // 16 threads per tile-row; each row is 16*CPAD_OUT halves, contiguous in NHWC.
  {
    const int row = tid >> 4, col = tid & 15;
    const uint4* srow = (const uint4*)&smem[row * 16 * CPAD_OUT];
    uint4* grow = (uint4*)&out[((size_t)(b * H + (y0 + row)) * W + x0) * CPAD_OUT];
#pragma unroll
    for (int j = 0; j < CPAD_OUT / 8; ++j) {
      int idx = col + j * 16;
      grow[idx] = srow[idx];
    }
  }
}

// ---------------------------------------------------------------------------
// pixel_unshuffle(x,4): fp32 [4][1][512][512] -> fp16 NHWC [4][128][128][32]
// One thread per site: 4x float4 gather, one 64B coalesced store.
// ---------------------------------------------------------------------------
__global__ void unshuffle4_k(const float* __restrict__ x, half_t* __restrict__ out, int n) {
  int i = blockIdx.x * blockDim.x + threadIdx.x;
  if (i >= n) return;  // n = B*128*128 sites
  int X = i & 127; int t = i >> 7;
  int Y = t & 127; int b = t >> 7;
  union { half_t h[32]; uint4 q[4]; } o;
#pragma unroll
  for (int r1 = 0; r1 < 4; ++r1) {
    const float4 row = *(const float4*)&x[((size_t)b * 512 + (Y * 4 + r1)) * 512 + X * 4];
    o.h[r1 * 4 + 0] = (half_t)row.x;
    o.h[r1 * 4 + 1] = (half_t)row.y;
    o.h[r1 * 4 + 2] = (half_t)row.z;
    o.h[r1 * 4 + 3] = (half_t)row.w;
  }
#pragma unroll
  for (int c = 16; c < 32; ++c) o.h[c] = (half_t)0.f;
  uint4* dst = (uint4*)&out[(size_t)i * 32];
#pragma unroll
  for (int k = 0; k < 4; ++k) dst[k] = o.q[k];
}

// pixel_shuffle(common,2): [4][128][128][64] -> [4][256][256][32] (16 real ch)
__global__ void shuffle2_k(const half_t* __restrict__ cm, half_t* __restrict__ out, int n) {
  int i = blockIdx.x * blockDim.x + threadIdx.x;
  if (i >= n) return;  // n = B*256*256 sites
  int X = i & 255; int t = i >> 8;
  int Y = t & 255; int b = t >> 8;
  const half_t* base =
      &cm[(((size_t)b * 128 + (Y >> 1)) * 128 + (X >> 1)) * 64 + (Y & 1) * 2 + (X & 1)];
  union { half_t h[32]; uint4 q[4]; } o;
#pragma unroll
  for (int c = 0; c < 16; ++c) o.h[c] = base[c * 4];
#pragma unroll
  for (int c = 16; c < 32; ++c) o.h[c] = (half_t)0.f;
  uint4* dst = (uint4*)&out[(size_t)i * 32];
#pragma unroll
  for (int k = 0; k < 4; ++k) dst[k] = o.q[k];
}

// pixel_shuffle(common,4): [4][128][128][64] -> [4][512][512][32] (4 real ch)
__global__ void shuffle4_k(const half_t* __restrict__ cm, half_t* __restrict__ out, int n) {
  int i = blockIdx.x * blockDim.x + threadIdx.x;
  if (i >= n) return;  // n = B*512*512 sites
  int X = i & 511; int t = i >> 9;
  int Y = t & 511; int b = t >> 9;
  const half_t* base =
      &cm[(((size_t)b * 128 + (Y >> 2)) * 128 + (X >> 2)) * 64 + (Y & 3) * 4 + (X & 3)];
  union { half_t h[32]; uint4 q[4]; } o;
#pragma unroll
  for (int c = 0; c < 4; ++c) o.h[c] = base[c * 16];
#pragma unroll
  for (int c = 4; c < 32; ++c) o.h[c] = (half_t)0.f;
  uint4* dst = (uint4*)&out[(size_t)i * 32];
#pragma unroll
  for (int k = 0; k < 4; ++k) dst[k] = o.q[k];
}

// ---------------------------------------------------------------------------
// KPN apply: e[b,y,x] = sum_{di,dj} filt[b,y,x,di*5+dj] * xpad[b, y*s+di, x*s+dj]
// (x zero-padded by 2). filt is fp16 NHWC with CPAD=32 (25 real channels).
// ---------------------------------------------------------------------------
__global__ void kpn_apply_k(const half_t* __restrict__ filt, const float* __restrict__ x,
                            float* __restrict__ e, int oh, int ow, int stride) {
  int X = blockIdx.x * blockDim.x + threadIdx.x;
  int Y = blockIdx.y, b = blockIdx.z;
  if (X >= ow) return;
  const half_t* f = &filt[(((size_t)b * oh + Y) * ow + X) * 32];
  float acc = 0.f;
#pragma unroll
  for (int di = 0; di < 5; ++di) {
    int sy = Y * stride + di - 2;
#pragma unroll
    for (int dj = 0; dj < 5; ++dj) {
      int sx = X * stride + dj - 2;
      float xv = (sy >= 0 && sy < 512 && sx >= 0 && sx < 512)
                     ? x[((size_t)b * 512 + sy) * 512 + sx] : 0.f;
      acc += (float)f[di * 5 + dj] * xv;
    }
  }
  e[((size_t)b * oh + Y) * ow + X] = acc;
}

// ---------------------------------------------------------------------------
// 2x bicubic upsample (separable 4-tap, edge-replicate), optional fused addend
// on the INPUT: out = up2(a + addend).
// ---------------------------------------------------------------------------
__global__ void up2_k(const float* __restrict__ a, const float* __restrict__ addend,
                      float* __restrict__ out, int h, int w) {
  int X = blockIdx.x * blockDim.x + threadIdx.x;
  int Y = blockIdx.y, b = blockIdx.z;
  int W2 = 2 * w;
  if (X >= W2) return;
  const float WE[4] = {-0.03515625f, 0.26171875f, 0.87890625f, -0.10546875f};
  int ix = X >> 1, pxo = X & 1;
  int iy = Y >> 1, pyo = Y & 1;
  float acc = 0.f;
#pragma unroll
  for (int jy = 0; jy < 4; ++jy) {
    float wy = pyo ? WE[3 - jy] : WE[jy];
    int sy = iy + jy - 2 + pyo;
    sy = sy < 0 ? 0 : (sy > h - 1 ? h - 1 : sy);
    float row = 0.f;
#pragma unroll
    for (int jx = 0; jx < 4; ++jx) {
      float wx = pxo ? WE[3 - jx] : WE[jx];
      int sx = ix + jx - 2 + pxo;
      sx = sx < 0 ? 0 : (sx > w - 1 ? w - 1 : sx);
      size_t idx = ((size_t)b * h + sy) * w + sx;
      float v = a[idx];
      if (addend) v += addend[idx];
      row += wx * v;
    }
    acc += wy * row;
  }
  out[((size_t)b * 2 * h + Y) * W2 + X] = acc;
}

// 5x5 single-channel conv, zero pad 2, weights = pyr_w (25 fp32).
__global__ void blur5_k(const float* __restrict__ in, const float* __restrict__ pw,
                        float* __restrict__ out, int h, int w) {
  int X = blockIdx.x * blockDim.x + threadIdx.x;
  int Y = blockIdx.y, b = blockIdx.z;
  if (X >= w) return;
  float acc = 0.f;
#pragma unroll
  for (int i = 0; i < 5; ++i) {
    int sy = Y + i - 2;
#pragma unroll
    for (int j = 0; j < 5; ++j) {
      int sx = X + j - 2;
      if (sy >= 0 && sy < h && sx >= 0 && sx < w)
        acc += pw[i * 5 + j] * in[((size_t)b * h + sy) * w + sx];
    }
  }
  out[((size_t)b * h + Y) * w + X] = acc;
}

// ---------------------------------------------------------------------------
// Host-side dispatch
// ---------------------------------------------------------------------------
static void launch_conv(const half_t* in, const half_t* wf, const float* bp, half_t* out,
                        int H, int W, int cpadIn, int ntOut, bool leaky, hipStream_t s) {
  dim3 grid(W / 16, H / 16, 4);
  if (cpadIn == 32) {
    conv3x3_wmma_k<32, 4, true, 64><<<grid, 256, 0, s>>>(in, wf, bp, out, H, W);
  } else if (ntOut == 2) {
    conv3x3_wmma_k<64, 2, true, 32><<<grid, 256, 0, s>>>(in, wf, bp, out, H, W);
  } else if (leaky) {
    conv3x3_wmma_k<64, 4, true, 64><<<grid, 256, 0, s>>>(in, wf, bp, out, H, W);
  } else {
    conv3x3_wmma_k<64, 4, false, 64><<<grid, 256, 0, s>>>(in, wf, bp, out, H, W);
  }
}

extern "C" void kernel_launch(void* const* d_in, const int* in_sizes, int n_in,
                              void* d_out, int out_size, void* d_ws, size_t ws_size,
                              hipStream_t stream) {
  (void)in_sizes; (void)n_in; (void)out_size; (void)ws_size;
  const float* x    = (const float*)d_in[0];
  const float* w1a  = (const float*)d_in[1];  const float* b1a  = (const float*)d_in[2];
  const float* w1b  = (const float*)d_in[3];  const float* b1b  = (const float*)d_in[4];
  const float* w1qa = (const float*)d_in[5];  const float* b1qa = (const float*)d_in[6];
  const float* w1qb = (const float*)d_in[7];  const float* b1qb = (const float*)d_in[8];
  const float* w1ha = (const float*)d_in[9];  const float* b1ha = (const float*)d_in[10];
  const float* w1hb = (const float*)d_in[11]; const float* b1hb = (const float*)d_in[12];
  const float* w1fa = (const float*)d_in[13]; const float* b1fa = (const float*)d_in[14];
  const float* w1fb = (const float*)d_in[15]; const float* b1fb = (const float*)d_in[16];
  const float* w2q  = (const float*)d_in[17]; const float* b2q  = (const float*)d_in[18];
  const float* w2h  = (const float*)d_in[19]; const float* b2h  = (const float*)d_in[20];
  const float* w2f  = (const float*)d_in[21]; const float* b2f  = (const float*)d_in[22];
  const float* stw  = (const float*)d_in[23];
  const float* stb  = (const float*)d_in[24];
  const float* pyrw = (const float*)d_in[25];

  char* ws = (char*)d_ws;
  size_t off = 0;
  auto alloc = [&](size_t bytes) -> char* {
    off = (off + 255) & ~(size_t)255;
    char* p = ws + off;
    off += bytes;
    return p;
  };

  const int B = 4;
  const int NLAYER = 45;
  const size_t FRAG_ELEMS = (size_t)9 * 2 * 4 * 32 * 16;  // max frag elems per layer

  half_t* fragBase = (half_t*)alloc(NLAYER * FRAG_ELEMS * sizeof(half_t));
  float*  biasBase = (float*)alloc((size_t)NLAYER * 64 * sizeof(float));
  auto F = [&](int i) { return fragBase + (size_t)i * FRAG_ELEMS; };
  auto Bp = [&](int i) { return biasBase + (size_t)i * 64; };

  // ---- pack all conv weights into WMMA fragment layout ----
  struct LayerDef { const float* w; const float* b; int cin, cout, KS, NT; };
  LayerDef L[NLAYER];
  L[0] = {w1a, b1a, 16, 64, 1, 4};
  for (int k = 0; k < 34; ++k)
    L[1 + k] = {stw + (size_t)k * 64 * 64 * 9, stb + (size_t)k * 64, 64, 64, 2, 4};
  L[35] = {w1b, b1b, 64, 64, 2, 4};
  L[36] = {w1qa, b1qa, 64, 64, 2, 4};
  L[37] = {w1qb, b1qb, 64, 64, 2, 4};
  L[38] = {w2q,  b2q,  64, 25, 2, 2};
  L[39] = {w1ha, b1ha, 16, 64, 1, 4};
  L[40] = {w1hb, b1hb, 64, 64, 2, 4};
  L[41] = {w2h,  b2h,  64, 25, 2, 2};
  L[42] = {w1fa, b1fa,  4, 64, 1, 4};
  L[43] = {w1fb, b1fb, 64, 64, 2, 4};
  L[44] = {w2f,  b2f,  64, 25, 2, 2};
  for (int i = 0; i < NLAYER; ++i) {
    int total = 9 * L[i].KS * L[i].NT * 32 * 16;
    pack_weights_k<<<(total + 255) / 256, 256, 0, stream>>>(
        L[i].w, L[i].b, F(i), Bp(i), L[i].cin, L[i].cout, L[i].KS, L[i].NT, total);
  }

  // ---- activation / intermediate buffers ----
  half_t* A0     = (half_t*)alloc((size_t)B * 128 * 128 * 32 * 2);
  half_t* C0     = (half_t*)alloc((size_t)B * 128 * 128 * 64 * 2);
  half_t* C1     = (half_t*)alloc((size_t)B * 128 * 128 * 64 * 2);
  half_t* common = (half_t*)alloc((size_t)B * 128 * 128 * 64 * 2);
  half_t* halfIn = (half_t*)alloc((size_t)B * 256 * 256 * 32 * 2);
  half_t* Hp0    = (half_t*)alloc((size_t)B * 256 * 256 * 64 * 2);
  half_t* Hp1    = (half_t*)alloc((size_t)B * 256 * 256 * 64 * 2);
  half_t* fullIn = (half_t*)alloc((size_t)B * 512 * 512 * 32 * 2);
  half_t* Fp0    = (half_t*)alloc((size_t)B * 512 * 512 * 64 * 2);
  half_t* Fp1    = (half_t*)alloc((size_t)B * 512 * 512 * 64 * 2);
  half_t* q32    = (half_t*)alloc((size_t)B * 128 * 128 * 32 * 2);
  half_t* h32    = (half_t*)alloc((size_t)B * 256 * 256 * 32 * 2);
  half_t* f32o   = (half_t*)alloc((size_t)B * 512 * 512 * 32 * 2);
  float* eq128   = (float*)alloc((size_t)B * 128 * 128 * 4);
  float* eh256   = (float*)alloc((size_t)B * 256 * 256 * 4);
  float* e512    = (float*)alloc((size_t)B * 512 * 512 * 4);
  float* t256a   = (float*)alloc((size_t)B * 256 * 256 * 4);
  float* t256b   = (float*)alloc((size_t)B * 256 * 256 * 4);
  float* t512a   = (float*)alloc((size_t)B * 512 * 512 * 4);
  float* t512b   = (float*)alloc((size_t)B * 512 * 512 * 4);

  // ---- trunk ----
  {
    int n = B * 128 * 128;
    unshuffle4_k<<<(n + 255) / 256, 256, 0, stream>>>(x, A0, n);
  }
  launch_conv(A0, F(0), Bp(0), C0, 128, 128, 32, 4, true, stream);        // w1a, leaky
  half_t* cur = C0; half_t* nxt = C1;
  launch_conv(cur, F(1), Bp(1), nxt, 128, 128, 64, 4, true, stream);      // stack_w[0], leaky
  { half_t* t = cur; cur = nxt; nxt = t; }
  for (int k = 0; k < 16; ++k) {
    launch_conv(cur, F(2 + 2 * k), Bp(2 + 2 * k), nxt, 128, 128, 64, 4, false, stream);
    { half_t* t = cur; cur = nxt; nxt = t; }
    launch_conv(cur, F(3 + 2 * k), Bp(3 + 2 * k), nxt, 128, 128, 64, 4, true, stream);
    { half_t* t = cur; cur = nxt; nxt = t; }
  }
  launch_conv(cur, F(34), Bp(34), nxt, 128, 128, 64, 4, false, stream);   // stack_w[33]
  { half_t* t = cur; cur = nxt; nxt = t; }
  launch_conv(cur, F(35), Bp(35), common, 128, 128, 64, 4, true, stream); // w1b, leaky

  // ---- quarter branch (128x128) ----
  launch_conv(common, F(36), Bp(36), C0, 128, 128, 64, 4, true, stream);
  launch_conv(C0,     F(37), Bp(37), C1, 128, 128, 64, 4, true, stream);
  launch_conv(C1,     F(38), Bp(38), q32, 128, 128, 64, 2, true, stream);

  // ---- half branch (256x256) ----
  {
    int n = B * 256 * 256;
    shuffle2_k<<<(n + 255) / 256, 256, 0, stream>>>(common, halfIn, n);
  }
  launch_conv(halfIn, F(39), Bp(39), Hp0, 256, 256, 32, 4, true, stream);
  launch_conv(Hp0,    F(40), Bp(40), Hp1, 256, 256, 64, 4, true, stream);
  launch_conv(Hp1,    F(41), Bp(41), h32, 256, 256, 64, 2, true, stream);

  // ---- full branch (512x512) ----
  {
    int n = B * 512 * 512;
    shuffle4_k<<<(n + 255) / 256, 256, 0, stream>>>(common, fullIn, n);
  }
  launch_conv(fullIn, F(42), Bp(42), Fp0, 512, 512, 32, 4, true, stream);
  launch_conv(Fp0,    F(43), Bp(43), Fp1, 512, 512, 64, 4, true, stream);
  launch_conv(Fp1,    F(44), Bp(44), f32o, 512, 512, 64, 2, true, stream);

  // ---- KPN apply at each scale ----
  kpn_apply_k<<<dim3((512 + 255) / 256, 512, B), 256, 0, stream>>>(f32o, x, e512, 512, 512, 1);
  kpn_apply_k<<<dim3((256 + 255) / 256, 256, B), 256, 0, stream>>>(h32, x, eh256, 256, 256, 2);
  kpn_apply_k<<<dim3((128 + 255) / 256, 128, B), 256, 0, stream>>>(q32, x, eq128, 128, 128, 4);

  // ---- bicubic pyramid: eq -> up2 -> blur; (+eh) -> up2 -> blur; (+e) -> up2 -> out
  up2_k<<<dim3((256 + 255) / 256, 256, B), 256, 0, stream>>>(eq128, nullptr, t256a, 128, 128);
  blur5_k<<<dim3((256 + 255) / 256, 256, B), 256, 0, stream>>>(t256a, pyrw, t256b, 256, 256);
  up2_k<<<dim3((512 + 255) / 256, 512, B), 256, 0, stream>>>(eh256, t256b, t512a, 256, 256);
  blur5_k<<<dim3((512 + 255) / 256, 512, B), 256, 0, stream>>>(t512a, pyrw, t512b, 512, 512);
  up2_k<<<dim3((1024 + 255) / 256, 1024, B), 256, 0, stream>>>(t512b, e512, (float*)d_out, 512, 512);
}
